// LinearSelfAttention_7816840478812
// MI455X (gfx1250) — matrix operand order repaired and checked by reading, not compiled, not run on hardware
//
#include <hip/hip_runtime.h>
#include <hip/hip_bf16.h>

typedef _Float16 h16;
typedef __attribute__((ext_vector_type(16))) _Float16 v16h;
typedef __attribute__((ext_vector_type(8)))  _Float16 v8h;
typedef __attribute__((ext_vector_type(8)))  float    v8f;
typedef __attribute__((ext_vector_type(4)))  int      v4i;

#define N_TOK   16384
#define CIN     512
#define NHEADS  8
#define HDIM    64

// padded LDS strides (halves); multiples of 8 halves => every row 16B aligned
#define SA 528   // for K=512 tiles
#define SN 144   // for K=128 tiles
#define SD 80    // for K=64  tiles

// ---- async global->LDS copy (CDNA5), guarded so compilation can't break ----
__device__ __forceinline__ void copy16_g2l(const h16* g, h16* l) {
#if __has_builtin(__builtin_amdgcn_global_load_async_to_lds_b128)
    __builtin_amdgcn_global_load_async_to_lds_b128(
        (__attribute__((address_space(1))) v4i*)(h16*)g,
        (__attribute__((address_space(3))) v4i*)l, 0, 0);
#else
    *(v8h*)l = *(const v8h*)g;
#endif
}
__device__ __forceinline__ void async_wait0() {
#if __has_builtin(__builtin_amdgcn_global_load_async_to_lds_b128)
#if __has_builtin(__builtin_amdgcn_s_wait_asynccnt)
    __builtin_amdgcn_s_wait_asynccnt(0);
#else
    asm volatile("s_wait_asynccnt 0" ::: "memory");
#endif
#endif
}

// Build a v16h WMMA fragment from two contiguous 16B LDS chunks.
__device__ __forceinline__ v16h frag16(const h16* p0, const h16* p1) {
    v8h lo = *(const v8h*)p0;
    v8h hi = *(const v8h*)p1;
    return __builtin_shufflevector(lo, hi, 0,1,2,3,4,5,6,7,8,9,10,11,12,13,14,15);
}

__device__ __forceinline__ v8f wmma_f16(v16h a, v16h b, v8f c) {
    return __builtin_amdgcn_wmma_f32_16x16x32_f16(false, a, false, b, (short)0, c, false, false);
}

// ---------------------------------------------------------------------------
// Kernel 1: qkv = xl @ w_qkv^T + b_qkv ; relu(q,k) ; store q,k (B,H,N,64) f16,
//           v (B,N,512) f16 with channel = h*64+c.
// grid = B * (N/64) = 1024 blocks, 128 threads (4 waves, 2x2 tiles each).
// ---------------------------------------------------------------------------
__global__ __launch_bounds__(128)
void qkv_kernel(const float* __restrict__ x, const float* __restrict__ wqkv,
                const float* __restrict__ bqkv,
                h16* __restrict__ qws, h16* __restrict__ kws, h16* __restrict__ vws) {
    __shared__ __align__(16) h16 As[64 * SA];   // x tile  [m][ch]
    __shared__ __align__(16) h16 Bs[64 * SA];   // w tile  [col][k]
    const int b  = blockIdx.x >> 8;
    const int n0 = (blockIdx.x & 255) * 64;
    const int t  = threadIdx.x;

    // stage x tile: As[m][ch] = x[b][ch][n0+m]  (coalesced float4 reads)
    {
        const float* xb = x + (size_t)b * CIN * N_TOK + n0;
        #pragma unroll 4
        for (int i = 0; i < 64; ++i) {
            int idx4 = t + i * 128;            // 8192 float4 total
            int ch   = idx4 >> 4;              // 16 float4 per channel
            int m4   = (idx4 & 15) * 4;
            const float4 v = *(const float4*)(xb + (size_t)ch * N_TOK + m4);
            As[(m4 + 0) * SA + ch] = (h16)v.x;
            As[(m4 + 1) * SA + ch] = (h16)v.y;
            As[(m4 + 2) * SA + ch] = (h16)v.z;
            As[(m4 + 3) * SA + ch] = (h16)v.w;
        }
    }
    const int wave = t >> 5, lane = t & 31;
    const int mrow = lane & 15;
    const int rbase = (lane >> 4) * 8;  // C-tile row base
    const int ka    = rbase;            // A-frag k offset
    const int kb16  = (lane >> 4) * 8;  // B-frag k offset
    const int tr0 = (wave >> 1) * 2;    // 2x2 tile footprint
    const int tc0 = (wave & 1) * 2;

    for (int jt = 0; jt < 24; ++jt) {
        const int col0 = jt * 64;
        __syncthreads();
        #pragma unroll 4
        for (int i = 0; i < 64; ++i) {
            int idx4 = t + i * 128;
            int col  = idx4 >> 7;              // 128 float4 per column
            int k4   = (idx4 & 127) * 4;
            const float4 v = *(const float4*)(wqkv + (size_t)(col0 + col) * CIN + k4);
            Bs[col * SA + k4 + 0] = (h16)v.x;
            Bs[col * SA + k4 + 1] = (h16)v.y;
            Bs[col * SA + k4 + 2] = (h16)v.z;
            Bs[col * SA + k4 + 3] = (h16)v.w;
        }
        __syncthreads();
        const h16* ar0 = &As[((tr0 + 0) * 16 + mrow) * SA];
        const h16* ar1 = &As[((tr0 + 1) * 16 + mrow) * SA];
        const h16* br0 = &Bs[((tc0 + 0) * 16 + mrow) * SA];
        const h16* br1 = &Bs[((tc0 + 1) * 16 + mrow) * SA];
        v8f acc00 = {}, acc01 = {}, acc10 = {}, acc11 = {};
        #pragma unroll
        for (int ks = 0; ks < 512; ks += 32) {
            v16h a0 = frag16(ar0 + ks + ka,   ar0 + ks + ka + 16);
            v16h a1 = frag16(ar1 + ks + ka,   ar1 + ks + ka + 16);
            v16h b0 = frag16(br0 + ks + kb16, br0 + ks + kb16 + 16);
            v16h b1 = frag16(br1 + ks + kb16, br1 + ks + kb16 + 16);
            acc00 = wmma_f16(a0, b0, acc00);
            acc01 = wmma_f16(a0, b1, acc01);
            acc10 = wmma_f16(a1, b0, acc10);
            acc11 = wmma_f16(a1, b1, acc11);
        }
        #pragma unroll
        for (int ri = 0; ri < 2; ++ri) {
            #pragma unroll
            for (int ci = 0; ci < 2; ++ci) {
                const v8f& acc = ri ? (ci ? acc11 : acc10) : (ci ? acc01 : acc00);
                const int c = (tc0 + ci) * 16 + mrow;
                const float bias = bqkv[col0 + c];
                #pragma unroll
                for (int j = 0; j < 8; ++j) {
                    float val = acc[j] + bias;
                    const int n = n0 + (tr0 + ri) * 16 + rbase + j;
                    if (jt < 16) {
                        val = val > 0.f ? val : 0.f;     // relu(q), relu(k)
                        const int h = jt & 7;
                        h16* dst = (jt < 8) ? qws : kws;
                        dst[((size_t)(b * NHEADS + h) * N_TOK + n) * HDIM + c] = (h16)val;
                    } else {
                        const int h = jt - 16;
                        vws[((size_t)b * N_TOK + n) * CIN + h * HDIM + c] = (h16)val;
                    }
                }
            }
        }
    }
}

// ---------------------------------------------------------------------------
// Kernel 2a: kq[b,h,c,d] += sum_n k[b,h,n,c] * q[b,h,n,d]  (split-K, atomics)
// grid = B*H*16 = 512 blocks, 128 threads.
// ---------------------------------------------------------------------------
__global__ __launch_bounds__(128)
void kq_kernel(const h16* __restrict__ qws, const h16* __restrict__ kws,
               float* __restrict__ kqf) {
    __shared__ __align__(16) h16 Kt[64 * SN];   // [c][n] (transposed)
    __shared__ __align__(16) h16 Qt[64 * SN];   // [d][n] (transposed)
    const int bh = blockIdx.x >> 4;
    const int s  = blockIdx.x & 15;
    const h16* kp = kws + (size_t)bh * N_TOK * HDIM;
    const h16* qp = qws + (size_t)bh * N_TOK * HDIM;
    const int t = threadIdx.x, wave = t >> 5, lane = t & 31;
    const int mrow = lane & 15, rbase = (lane >> 4) * 8;
    const int ka = rbase, kb16 = (lane >> 4) * 8;
    const int tr0 = (wave >> 1) * 2, tc0 = (wave & 1) * 2;

    v8f acc00 = {}, acc01 = {}, acc10 = {}, acc11 = {};
    for (int chunk = 0; chunk < 8; ++chunk) {
        const int nbase = s * 1024 + chunk * 128;
        __syncthreads();
        #pragma unroll 4
        for (int i = 0; i < 64; ++i) {
            int idx = t + i * 128;             // 8192 halves each
            int n = idx >> 6, c = idx & 63;
            Kt[c * SN + n] = kp[(size_t)(nbase + n) * HDIM + c];
            Qt[c * SN + n] = qp[(size_t)(nbase + n) * HDIM + c];
        }
        __syncthreads();
        const h16* ar0 = &Kt[((tr0 + 0) * 16 + mrow) * SN];
        const h16* ar1 = &Kt[((tr0 + 1) * 16 + mrow) * SN];
        const h16* br0 = &Qt[((tc0 + 0) * 16 + mrow) * SN];
        const h16* br1 = &Qt[((tc0 + 1) * 16 + mrow) * SN];
        #pragma unroll
        for (int ks = 0; ks < 128; ks += 32) {
            v16h a0 = frag16(ar0 + ks + ka,   ar0 + ks + ka + 16);
            v16h a1 = frag16(ar1 + ks + ka,   ar1 + ks + ka + 16);
            v16h b0 = frag16(br0 + ks + kb16, br0 + ks + kb16 + 16);
            v16h b1 = frag16(br1 + ks + kb16, br1 + ks + kb16 + 16);
            acc00 = wmma_f16(a0, b0, acc00);
            acc01 = wmma_f16(a0, b1, acc01);
            acc10 = wmma_f16(a1, b0, acc10);
            acc11 = wmma_f16(a1, b1, acc11);
        }
    }
    #pragma unroll
    for (int ri = 0; ri < 2; ++ri) {
        #pragma unroll
        for (int ci = 0; ci < 2; ++ci) {
            const v8f& acc = ri ? (ci ? acc11 : acc10) : (ci ? acc01 : acc00);
            int d = (tc0 + ci) * 16 + mrow;
            #pragma unroll
            for (int j = 0; j < 8; ++j) {
                int c = (tr0 + ri) * 16 + rbase + j;
                atomicAdd(&kqf[((size_t)bh * 64 + c) * 64 + d], acc[j]);
            }
        }
    }
}

// ---------------------------------------------------------------------------
// Kernel 2b: W2[b,o,h*64+c] = sum_d w_out[o,d*8+h] * kq[b,h,c,d]/norm[b,h,d]
// grid = B*H = 32 blocks, 128 threads.
// ---------------------------------------------------------------------------
__global__ __launch_bounds__(128)
void w2_kernel(const float* __restrict__ kqf, const float* __restrict__ wout,
               h16* __restrict__ w2) {
    __shared__ float kqs[64 * 64];
    __shared__ float invn[64];
    __shared__ __align__(16) h16 Bn[64 * SD];    // [c][d] normalized kq^T
    __shared__ __align__(16) h16 As2[64 * SD];   // [o][d] permuted w_out
    const int bh = blockIdx.x;
    const int b = bh >> 3, h = bh & 7;
    const int t = threadIdx.x;
    #pragma unroll
    for (int i = 0; i < 32; ++i)
        kqs[t + i * 128] = kqf[(size_t)bh * 4096 + t + i * 128];
    __syncthreads();
    if (t < 64) {
        float sum = 0.f;
        for (int c = 0; c < 64; ++c) sum += kqs[c * 64 + t];
        invn[t] = 1.0f / (sum + 1e-6f);
    }
    __syncthreads();
    #pragma unroll
    for (int i = 0; i < 32; ++i) {
        int idx = t + i * 128;
        int c = idx >> 6, d = idx & 63;
        Bn[c * SD + d] = (h16)(kqs[c * 64 + d] * invn[d]);
    }
    const int wave = t >> 5, lane = t & 31;
    const int mrow = lane & 15, rbase = (lane >> 4) * 8;
    const int ka = rbase, kb16 = (lane >> 4) * 8;
    const int tr0 = (wave >> 1) * 2, tc0 = (wave & 1) * 2;
    for (int ot = 0; ot < 8; ++ot) {
        const int o0 = ot * 64;
        __syncthreads();
        #pragma unroll
        for (int i = 0; i < 32; ++i) {
            int idx = t + i * 128;
            int o = idx >> 6, d = idx & 63;
            As2[o * SD + d] = (h16)wout[(size_t)(o0 + o) * CIN + d * 8 + h];
        }
        __syncthreads();
        const h16* ar0 = &As2[((tr0 + 0) * 16 + mrow) * SD];
        const h16* ar1 = &As2[((tr0 + 1) * 16 + mrow) * SD];
        const h16* br0 = &Bn[((tc0 + 0) * 16 + mrow) * SD];
        const h16* br1 = &Bn[((tc0 + 1) * 16 + mrow) * SD];
        v8f acc00 = {}, acc01 = {}, acc10 = {}, acc11 = {};
        #pragma unroll
        for (int ks = 0; ks < 64; ks += 32) {
            v16h a0 = frag16(ar0 + ks + ka,   ar0 + ks + ka + 16);
            v16h a1 = frag16(ar1 + ks + ka,   ar1 + ks + ka + 16);
            v16h b0 = frag16(br0 + ks + kb16, br0 + ks + kb16 + 16);
            v16h b1 = frag16(br1 + ks + kb16, br1 + ks + kb16 + 16);
            acc00 = wmma_f16(a0, b0, acc00);
            acc01 = wmma_f16(a0, b1, acc01);
            acc10 = wmma_f16(a1, b0, acc10);
            acc11 = wmma_f16(a1, b1, acc11);
        }
        #pragma unroll
        for (int ri = 0; ri < 2; ++ri) {
            #pragma unroll
            for (int ci = 0; ci < 2; ++ci) {
                const v8f& acc = ri ? (ci ? acc11 : acc10) : (ci ? acc01 : acc00);
                int c = (tc0 + ci) * 16 + mrow;
                #pragma unroll
                for (int j = 0; j < 8; ++j) {
                    int o = o0 + (tr0 + ri) * 16 + rbase + j;
                    w2[((size_t)b * CIN + o) * CIN + h * HDIM + c] = (h16)acc[j];
                }
            }
        }
    }
}

// ---------------------------------------------------------------------------
// Kernel 3: out[b,o,n] = sum_ch v[b,n,ch] * W2[b,o,ch] + b_out[o]
// grid = B * (N/64) = 1024 blocks, 128 threads. Async global->LDS staging.
// ---------------------------------------------------------------------------
__global__ __launch_bounds__(128)
void out_kernel(const h16* __restrict__ vws, const h16* __restrict__ w2,
                const float* __restrict__ bout, float* __restrict__ out) {
    __shared__ __align__(16) h16 Vs[64 * SA];   // [n][ch]
    __shared__ __align__(16) h16 Ws[64 * SA];   // [o][ch]
    const int b  = blockIdx.x >> 8;
    const int n0 = (blockIdx.x & 255) * 64;
    const int t  = threadIdx.x;
    {
        const h16* vp = vws + ((size_t)b * N_TOK + n0) * CIN;
        #pragma unroll 4
        for (int i = 0; i < 32; ++i) {
            int idx = t + i * 128;             // 4096 8-half chunks
            int n = idx >> 6, c8 = (idx & 63) * 8;
            copy16_g2l(vp + (size_t)n * CIN + c8, &Vs[n * SA + c8]);
        }
        async_wait0();
    }
    const int wave = t >> 5, lane = t & 31;
    const int mrow = lane & 15, rbase = (lane >> 4) * 8;
    const int ka = rbase, kb16 = (lane >> 4) * 8;
    const int tr0 = (wave >> 1) * 2, tc0 = (wave & 1) * 2;
    for (int ot = 0; ot < 8; ++ot) {
        const int o0 = ot * 64;
        __syncthreads();
        const h16* wp = w2 + ((size_t)b * CIN + o0) * CIN;
        #pragma unroll 4
        for (int i = 0; i < 32; ++i) {
            int idx = t + i * 128;
            int o = idx >> 6, c8 = (idx & 63) * 8;
            copy16_g2l(wp + (size_t)o * CIN + c8, &Ws[o * SA + c8]);
        }
        async_wait0();
        __syncthreads();
        const h16* ar0 = &Vs[((tr0 + 0) * 16 + mrow) * SA];
        const h16* ar1 = &Vs[((tr0 + 1) * 16 + mrow) * SA];
        const h16* br0 = &Ws[((tc0 + 0) * 16 + mrow) * SA];
        const h16* br1 = &Ws[((tc0 + 1) * 16 + mrow) * SA];
        v8f acc00 = {}, acc01 = {}, acc10 = {}, acc11 = {};
        #pragma unroll
        for (int ks = 0; ks < 512; ks += 32) {
            v16h a0 = frag16(ar0 + ks + ka,   ar0 + ks + ka + 16);
            v16h a1 = frag16(ar1 + ks + ka,   ar1 + ks + ka + 16);
            v16h b0 = frag16(br0 + ks + kb16, br0 + ks + kb16 + 16);
            v16h b1 = frag16(br1 + ks + kb16, br1 + ks + kb16 + 16);
            acc00 = wmma_f16(a0, b0, acc00);
            acc01 = wmma_f16(a0, b1, acc01);
            acc10 = wmma_f16(a1, b0, acc10);
            acc11 = wmma_f16(a1, b1, acc11);
        }
        #pragma unroll
        for (int ri = 0; ri < 2; ++ri) {
            #pragma unroll
            for (int ci = 0; ci < 2; ++ci) {
                const v8f& acc = ri ? (ci ? acc11 : acc10) : (ci ? acc01 : acc00);
                int o = o0 + (tc0 + ci) * 16 + mrow;
                float bias = bout[o];
                v8f res;
                #pragma unroll
                for (int j = 0; j < 8; ++j) res[j] = acc[j] + bias;
                // rows of D are consecutive n -> one 32B store per lane
                float* dst = out + ((size_t)(b * CIN + o)) * N_TOK
                           + n0 + (tr0 + ri) * 16 + rbase;
                *(v8f*)dst = res;
            }
        }
    }
}

// ---------------------------------------------------------------------------
extern "C" void kernel_launch(void* const* d_in, const int* in_sizes, int n_in,
                              void* d_out, int out_size, void* d_ws, size_t ws_size,
                              hipStream_t stream) {
    (void)in_sizes; (void)n_in; (void)out_size; (void)ws_size;
    const float* x    = (const float*)d_in[0];
    const float* wqkv = (const float*)d_in[1];
    const float* bqkv = (const float*)d_in[2];
    const float* wout = (const float*)d_in[3];
    const float* bout = (const float*)d_in[4];
    float* out = (float*)d_out;

    char* ws = (char*)d_ws;
    h16*   qws = (h16*)(ws);                               // 64 MiB
    h16*   kws = (h16*)(ws + (size_t)67108864);            // 64 MiB
    h16*   vws = (h16*)(ws + (size_t)134217728);           // 64 MiB
    float* kqf = (float*)(ws + (size_t)201326592);         // 512 KiB
    h16*   w2  = (h16*)(ws + (size_t)201326592 + 524288);  // 2 MiB

    (void)hipMemsetAsync(kqf, 0, (size_t)32 * 64 * 64 * sizeof(float), stream);
    qkv_kernel<<<1024, 128, 0, stream>>>(x, wqkv, bqkv, qws, kws, vws);
    kq_kernel <<< 512, 128, 0, stream>>>(qws, kws, kqf);
    w2_kernel <<<  32, 128, 0, stream>>>(kqf, wout, w2);
    out_kernel<<<1024, 128, 0, stream>>>(vws, w2, bout, out);
}
